// MambaEncoder_48859547959719
// MI455X (gfx1250) — matrix operand order repaired and checked by reading, not compiled, not run on hardware
//
#include <hip/hip_runtime.h>
#include <hip/hip_bf16.h>
#include <math.h>

#define B_   32
#define L_   4096
#define DM_  64
#define DS_  64
#define DC_  4
#define DI_  128
#define NXZ  256   // 2*DI
#define LT   128   // L-rows per front tile
#define ZROW 144   // padded z tile rows (3-halo + 128 + tail pad)
#define XST  152   // column stride of column-major sX (aligned, bank-skewed)

typedef __bf16 bf16_t;
typedef __attribute__((ext_vector_type(16))) __bf16 v16bf;
typedef __attribute__((ext_vector_type(8)))  __bf16 v8bf;
typedef __attribute__((ext_vector_type(8)))  float  v8f;

// ---- gfx1250 async LDS copy (ASYNCcnt path), guarded by __has_builtin ----
#if __has_builtin(__builtin_amdgcn_global_load_async_to_lds_b128) && \
    __has_builtin(__builtin_amdgcn_s_wait_asynccnt)
#define ASYNC_LDS 1
typedef int v4i_t __attribute__((vector_size(16)));
typedef __attribute__((address_space(1))) v4i_t gv4i;
typedef __attribute__((address_space(3))) v4i_t lv4i;
__device__ __forceinline__ void cp_async16(void* dst_lds, const void* src_gbl) {
  gv4i* s = (gv4i*)(unsigned long long)src_gbl;
  lv4i* d = (lv4i*)(unsigned)(unsigned long long)dst_lds;
  __builtin_amdgcn_global_load_async_to_lds_b128(s, d, 0, 0);
}
#endif

// ---- WMMA fragment helpers (CDNA5 ISA 7.12.2 16-bit layouts, wave32) ----
// A fragment: per-lane elements 0..7 = K[kh..kh+7], 8..15 = K[16+kh..16+kh+7]
// (kh = 8*(lane>>4)) => two contiguous 16B LDS loads (ds_load_b128).
__device__ __forceinline__ v16bf frag_a(const bf16_t* p, int lda) {
  int lane = threadIdx.x & 31;
  const bf16_t* row = p + (lane & 15) * lda + ((lane >> 4) << 3);
  v8bf lo = *(const v8bf*)(row);
  v8bf hi = *(const v8bf*)(row + 16);
  return __builtin_shufflevector(lo, hi, 0, 1, 2, 3, 4, 5, 6, 7,
                                 8, 9, 10, 11, 12, 13, 14, 15);
}
// B operand is stored PRE-SWIZZLED in LDS: per (ktile,ntile) 32x16 block,
// lane-major, 16 bf16 (32B) contiguous per lane => two ds_load_b128.
__device__ __forceinline__ v16bf frag_b_sw(const bf16_t* blockp) {
  const bf16_t* q = blockp + (threadIdx.x & 31) * 16;
  v8bf lo = *(const v8bf*)(q);
  v8bf hi = *(const v8bf*)(q + 8);
  return __builtin_shufflevector(lo, hi, 0, 1, 2, 3, 4, 5, 6, 7,
                                 8, 9, 10, 11, 12, 13, 14, 15);
}
// Swizzled B-store index for element (k, n) with NT n-tiles total.
__device__ __forceinline__ int bswz(int k, int n, int NT) {
  return (((k >> 5) * NT + (n >> 4)) << 9) + ((((k >> 3) & 1) * 16 + (n & 15)) << 4)
         + (((k >> 4) & 1) << 3) + (k & 7);
}
__device__ __forceinline__ v8f wmma_bf16(v16bf a, v16bf b, v8f c) {
  return __builtin_amdgcn_wmma_f32_16x16x32_bf16(false, a, false, b, (short)0, c,
                                                 false, false);
}
__device__ __forceinline__ float sigmoidf_(float x) { return 1.f / (1.f + __expf(-x)); }

// =====================================================================
// Kernel 1: embed -> in_proj (WMMA) -> causal conv + SiLU -> x_proj (WMMA)
//           -> dt_proj + softplus.  One block per (batch, 128-row tile).
// 144-row halo so the conv never crosses tiles. All fragment stores are
// wave-uniform; x/dtr tiles stored column-major so each lane's 8-row
// fragment column is contiguous (one ds_store_b128 for x, two for dtr).
// =====================================================================
__global__ __launch_bounds__(256) void mamba_front(
    const float* __restrict__ reads, const float* __restrict__ exp_w,
    const float* __restrict__ exp_b, const float* __restrict__ in_proj_w,
    const float* __restrict__ conv_w, const float* __restrict__ conv_b,
    const float* __restrict__ x_proj_w, const float* __restrict__ dt_proj_w,
    const float* __restrict__ dt_proj_b,
    float* __restrict__ z_o, float* __restrict__ xc_o, float* __restrict__ dt_o,
    float* __restrict__ Bm_o, float* __restrict__ Cm_o)
{
  // Hand-packed LDS arena, phase-based reuse (peak ~114 KB < 320 KB/WG).
  __shared__ __align__(16) char smem[116736];
  bf16_t* sH    = (bf16_t*)(smem);           // [A/B] 144x64  bf16    18432 B
  bf16_t* sWxz  = (bf16_t*)(smem + 18432);   // [A/B] swizzled 64x256 32768 B
  bf16_t* sX    = (bf16_t*)(smem + 69632);   // [B/C] col-major 128xXST 38912 B
  bf16_t* sXC   = (bf16_t*)(smem);           // [C/D] 128x128 bf16 (over sH/sWxz)
  bf16_t* sWxp  = (bf16_t*)(smem + 32768);   // [C/D] swizzled 128x144 36864 B
  float*  sDtrP = (float*)(smem + 108544);   // [D/E] col-major 16x128 8192 B

  const int b    = blockIdx.y;
  const int l0   = blockIdx.x * LT;
  const int tid  = threadIdx.x;
  const int wave = tid >> 5;
  const int lane = tid & 31;

  // ---- Phase A: h = reads @ exp_w.T + exp_b (144 halo rows, K=4 VALU) ----
  for (int i = tid; i < 144 * DM_; i += 256) {
    int r = i >> 6, c = i & 63;
    int l = l0 - 3 + r;
    float h = 0.f;                       // zero rows => zero x => causal padding
    if (l >= 0 && r < 131) {
      const float* rd = reads + ((size_t)b * L_ + l) * 4;
      h = exp_b[c];
#pragma unroll
      for (int j = 0; j < 4; ++j) h += rd[j] * exp_w[c * 4 + j];
    }
    sH[i] = (bf16_t)h;
  }
  for (int i = tid; i < DM_ * NXZ; i += 256) {     // in_proj_w -> swizzled B
    int k = i & 63, n = i >> 6;
    sWxz[bswz(k, n, 16)] = (bf16_t)in_proj_w[n * DM_ + k];
  }
  __syncthreads();

  // ---- Phase B: xz = h @ Wxz (144x256, K=64). Split x-half / z-half ----
  for (int idx = wave; idx < 72; idx += 8) {       // x half: 9m x 8n tiles
    int mt = idx >> 3, nt = idx & 7;
    v8f acc = {};
#pragma unroll
    for (int kk = 0; kk < 2; ++kk) {
      v16bf a  = frag_a(sH + (mt * 16) * DM_ + kk * 32, DM_);
      v16bf bb = frag_b_sw(sWxz + ((kk * 16 + nt) << 9));
      acc = wmma_bf16(a, bb, acc);
    }
    int n  = nt * 16 + (lane & 15);
    int mb = mt * 16 + ((lane >> 4) << 3);
    v8bf pk;                             // column-major: 8 rows contiguous
#pragma unroll
    for (int r = 0; r < 8; ++r) pk[r] = (bf16_t)acc[r];
    *(v8bf*)(sX + n * XST + mb) = pk;    // single ds_store_b128
  }
  {
    const size_t zb = (size_t)(b * 32 + blockIdx.x) * ZROW * DI_;
    for (int idx = wave; idx < 72; idx += 8) {     // z half: 9m x 8n tiles
      int mt = idx >> 3, nt = idx & 7;
      v8f acc = {};
#pragma unroll
      for (int kk = 0; kk < 2; ++kk) {
        v16bf a  = frag_a(sH + (mt * 16) * DM_ + kk * 32, DM_);
        v16bf bb = frag_b_sw(sWxz + ((kk * 16 + 8 + nt) << 9));
        acc = wmma_bf16(a, bb, acc);
      }
      int zc = nt * 16 + (lane & 15);
      int mb = mt * 16 + ((lane >> 4) << 3);
#pragma unroll
      for (int r = 0; r < 8; ++r)                  // lane-coalesced row-major
        z_o[zb + (size_t)(mb + r) * DI_ + zc] = acc[r];
    }
  }
  __syncthreads();

  // ---- Phase C: depthwise causal conv (DC=4) + SiLU; stage x_proj weights ----
  for (int i = tid; i < LT * DI_; i += 256) {
    int m = i >> 7, d = i & 127;
    float acc = conv_b[d];
#pragma unroll
    for (int k = 0; k < DC_; ++k)
      acc += (float)sX[d * XST + (m + k)] * conv_w[d * DC_ + k];
    float xs = acc * sigmoidf_(acc);
    sXC[m * DI_ + d] = (bf16_t)xs;
    xc_o[((size_t)b * L_ + (l0 + m)) * DI_ + d] = xs;
  }
  // x_proj_w with padded columns: n' 0..15 -> dtr(0..3)+zeros, 16..79 -> B rows
  // (src n'-12), 80..143 -> C rows (src n'-12).  Swizzled B layout, NT=9.
  for (int i = tid; i < DI_ * 144; i += 256) {
    int k = i & 127, np = i >> 7;
    float w = 0.f;
    if (np < 4)        w = x_proj_w[np * DI_ + k];
    else if (np >= 16) w = x_proj_w[(np - 12) * DI_ + k];
    sWxp[bswz(k, np, 9)] = (bf16_t)w;
  }
  __syncthreads();

  // ---- Phase D: dbl = xc @ Wxp (128x144 padded, K=128). Uniform stores ----
  for (int idx = wave; idx < 72; idx += 8) {       // 8m x 9n tiles
    int mt = idx / 9, nt = idx - mt * 9;
    v8f acc = {};
#pragma unroll
    for (int kk = 0; kk < 4; ++kk) {
      v16bf a  = frag_a(sXC + (mt * 16) * DI_ + kk * 32, DI_);
      v16bf bb = frag_b_sw(sWxp + ((kk * 9 + nt) << 9));
      acc = wmma_bf16(a, bb, acc);
    }
    int nn = lane & 15;
    int mb = mt * 16 + ((lane >> 4) << 3);
    if (nt == 0) {                                 // dtr tile, col-major LDS
      *(float4*)(sDtrP + nn * 128 + mb)     = make_float4(acc[0], acc[1], acc[2], acc[3]);
      *(float4*)(sDtrP + nn * 128 + mb + 4) = make_float4(acc[4], acc[5], acc[6], acc[7]);
    } else if (nt <= 4) {                          // B tile (lane-coalesced)
      int col = (nt - 1) * 16 + nn;
#pragma unroll
      for (int r = 0; r < 8; ++r)
        Bm_o[((size_t)b * L_ + (l0 + mb + r)) * DS_ + col] = acc[r];
    } else {                                       // C tile (lane-coalesced)
      int col = (nt - 5) * 16 + nn;
#pragma unroll
      for (int r = 0; r < 8; ++r)
        Cm_o[((size_t)b * L_ + (l0 + mb + r)) * DS_ + col] = acc[r];
    }
  }
  __syncthreads();

  // ---- Phase E: dt = softplus(dtr @ dt_proj_w.T + dt_proj_b) (K=4 VALU) ----
  for (int i = tid; i < LT * DI_; i += 256) {
    int m = i >> 7, d = i & 127;
    float v = dt_proj_b[d];
#pragma unroll
    for (int j = 0; j < 4; ++j) v += sDtrP[j * 128 + m] * dt_proj_w[d * 4 + j];
    float sp = (v > 20.f) ? v : log1pf(__expf(v));
    dt_o[((size_t)b * L_ + (l0 + m)) * DI_ + d] = sp;
  }
}

// =====================================================================
// Kernel 2: selective scan, fused gating + linear-pooling reduction.
// One block per batch; 2 threads per channel, 32 states each in VGPRs.
// Staging uses gfx1250 GLOBAL_LOAD_ASYNC_TO_LDS_B128 (ASYNCcnt) when the
// builtin exists, else float4 load/store pairs.
// =====================================================================
__global__ __launch_bounds__(256) void mamba_scan(
    const float* __restrict__ xc, const float* __restrict__ dt,
    const float* __restrict__ z,  const float* __restrict__ Bm,
    const float* __restrict__ Cm, const float* __restrict__ A_log,
    const float* __restrict__ Dp, float* __restrict__ pooled)
{
  const int b   = blockIdx.x;
  const int tid = threadIdx.x;
  const int d   = tid >> 1;             // channel
  const int sh  = (tid & 1) * 32;       // state half owned by this thread

  float A[32], hst[32];
#pragma unroll
  for (int s = 0; s < 32; ++s) { A[s] = -__expf(A_log[d * DS_ + sh + s]); hst[s] = 0.f; }
  const float Dd = Dp[d];

  __shared__ __align__(16) float sB[16 * DS_];   //  4 KB
  __shared__ __align__(16) float sC[16 * DS_];   //  4 KB
  __shared__ __align__(16) float sDT[16 * DI_];  //  8 KB
  __shared__ __align__(16) float sXV[16 * DI_];  //  8 KB
  __shared__ __align__(16) float sZV[16 * DI_];  //  8 KB

  float ysum = 0.f, ylast = 0.f;
  const size_t base = (size_t)b * L_;

  for (int t0 = 0; t0 < L_; t0 += 16) {
    if (t0 + 16 < L_) {                 // gfx1250 global_prefetch of next chunk
      __builtin_prefetch(Bm + (base + t0 + 16) * DS_, 0, 1);
      __builtin_prefetch(Cm + (base + t0 + 16) * DS_, 0, 1);
    }
    // z lives in a 144-row padded per-tile layout (rows 3..130 are real).
    const size_t ztb = ((size_t)(b * 32 + (t0 >> 7)) * ZROW + (t0 & 127) + 3) * DI_;
    const float4* gB = (const float4*)(Bm + (base + t0) * DS_);
    const float4* gC = (const float4*)(Cm + (base + t0) * DS_);
    const float4* gD = (const float4*)(dt + (base + t0) * DI_);
    const float4* gX = (const float4*)(xc + (base + t0) * DI_);
    const float4* gZ = (const float4*)(z + ztb);
    __syncthreads();                    // previous chunk fully consumed
#ifdef ASYNC_LDS
    cp_async16((float4*)sB + tid,        gB + tid);        // 256 lanes x 16B
    cp_async16((float4*)sC + tid,        gC + tid);
    cp_async16((float4*)sDT + tid,       gD + tid);        // 512 x 16B (2 each)
    cp_async16((float4*)sDT + tid + 256, gD + tid + 256);
    cp_async16((float4*)sXV + tid,       gX + tid);
    cp_async16((float4*)sXV + tid + 256, gX + tid + 256);
    cp_async16((float4*)sZV + tid,       gZ + tid);
    cp_async16((float4*)sZV + tid + 256, gZ + tid + 256);
    __builtin_amdgcn_s_wait_asynccnt(0);
#else
    ((float4*)sB)[tid] = gB[tid];
    ((float4*)sC)[tid] = gC[tid];
    ((float4*)sDT)[tid]       = gD[tid];
    ((float4*)sDT)[tid + 256] = gD[tid + 256];
    ((float4*)sXV)[tid]       = gX[tid];
    ((float4*)sXV)[tid + 256] = gX[tid + 256];
    ((float4*)sZV)[tid]       = gZ[tid];
    ((float4*)sZV)[tid + 256] = gZ[tid + 256];
#endif
    __syncthreads();
#pragma unroll 4
    for (int tt = 0; tt < 16; ++tt) {
      const float dtv = sDT[tt * DI_ + d];
      const float xv  = sXV[tt * DI_ + d];
      const float zv  = sZV[tt * DI_ + d];
      const float dx  = dtv * xv;
      const float4* pB = (const float4*)(sB + tt * DS_ + sh);
      const float4* pC = (const float4*)(sC + tt * DS_ + sh);
      float y = 0.f;
#pragma unroll
      for (int q = 0; q < 8; ++q) {     // 32 states as 8 x float4 (b128)
        float4 b4 = pB[q];
        float4 c4 = pC[q];
        float h0 = hst[4 * q + 0] * __expf(dtv * A[4 * q + 0]) + dx * b4.x;
        float h1 = hst[4 * q + 1] * __expf(dtv * A[4 * q + 1]) + dx * b4.y;
        float h2 = hst[4 * q + 2] * __expf(dtv * A[4 * q + 2]) + dx * b4.z;
        float h3 = hst[4 * q + 3] * __expf(dtv * A[4 * q + 3]) + dx * b4.w;
        hst[4 * q + 0] = h0; hst[4 * q + 1] = h1;
        hst[4 * q + 2] = h2; hst[4 * q + 3] = h3;
        y += h0 * c4.x + h1 * c4.y + h2 * c4.z + h3 * c4.w;
      }
      y += __shfl_xor(y, 1, 32);        // combine the two state-halves (wave32)
      float yt = (y + xv * Dd) * (zv * sigmoidf_(zv));
      ysum += yt;
      ylast = yt;                       // final executed step is t = L-1
    }
  }
  if ((tid & 1) == 0) {                 // rows 0..31 = mean, rows 32..63 = last
    pooled[b * DI_ + d]        = ysum * (1.f / (float)L_);
    pooled[(32 + b) * DI_ + d] = ylast;
  }
}

// =====================================================================
// Kernel 3: [ybar; ylast] (64x128) @ out_proj_w.T (128x64) via WMMA.
// =====================================================================
__global__ __launch_bounds__(128) void mamba_out(
    const float* __restrict__ pooled, const float* __restrict__ out_proj_w,
    float* __restrict__ out)
{
  __shared__ __align__(16) bf16_t sA[64 * DI_];   // 16 KB, row-major
  __shared__ __align__(16) bf16_t sW[DI_ * DM_];  // 16 KB, swizzled (NT=4)
  const int tid = threadIdx.x, wave = tid >> 5, lane = tid & 31;

  for (int i = tid; i < 64 * DI_; i += 128) sA[i] = (bf16_t)pooled[i];
  for (int i = tid; i < DI_ * DM_; i += 128) {
    int k = i >> 6, n = i & 63;
    sW[bswz(k, n, 4)] = (bf16_t)out_proj_w[n * DI_ + k];
  }
  __syncthreads();

  for (int idx = wave; idx < 16; idx += 4) {
    int mt = idx >> 2, nt = idx & 3;
    v8f acc = {};
#pragma unroll
    for (int kk = 0; kk < 4; ++kk) {
      v16bf a  = frag_a(sA + (mt * 16) * DI_ + kk * 32, DI_);
      v16bf bb = frag_b_sw(sW + ((kk * 4 + nt) << 9));
      acc = wmma_bf16(a, bb, acc);
    }
    int n = nt * 16 + (lane & 15);
#pragma unroll
    for (int r = 0; r < 8; ++r) {
      int row = mt * 16 + ((lane >> 4) << 3) + r;
      // rows 0..31 = mean (cols 0..63), rows 32..63 = last (cols 64..127)
      out[(row & 31) * 128 + ((row >> 5) << 6) + n] = acc[r];
    }
  }
}

// =====================================================================
extern "C" void kernel_launch(void* const* d_in, const int* in_sizes, int n_in,
                              void* d_out, int out_size, void* d_ws, size_t ws_size,
                              hipStream_t stream) {
  (void)in_sizes; (void)n_in; (void)out_size; (void)ws_size;
  const float* reads      = (const float*)d_in[0];
  const float* exp_w      = (const float*)d_in[1];
  const float* exp_b      = (const float*)d_in[2];
  const float* in_proj_w  = (const float*)d_in[3];
  const float* conv_w     = (const float*)d_in[4];
  const float* conv_b     = (const float*)d_in[5];
  const float* x_proj_w   = (const float*)d_in[6];
  const float* dt_proj_w  = (const float*)d_in[7];
  const float* dt_proj_b  = (const float*)d_in[8];
  const float* A_log      = (const float*)d_in[9];
  const float* Dp         = (const float*)d_in[10];
  const float* out_proj_w = (const float*)d_in[11];

  float* ws = (float*)d_ws;
  const size_t nBLDI = (size_t)B_ * L_ * DI_;            // 16.7M floats
  const size_t nBLDS = (size_t)B_ * L_ * DS_;            // 8.4M floats
  const size_t nZPAD = (size_t)B_ * 32 * ZROW * DI_;     // 18.9M floats (padded z)
  float* z_ws   = ws;
  float* xc_ws  = z_ws  + nZPAD;
  float* dt_ws  = xc_ws + nBLDI;
  float* Bm_ws  = dt_ws + nBLDI;
  float* Cm_ws  = Bm_ws + nBLDS;
  float* pooled = Cm_ws + nBLDS;                         // 2*B*DI floats

  dim3 g1(L_ / LT, B_);
  mamba_front<<<g1, 256, 0, stream>>>(reads, exp_w, exp_b, in_proj_w, conv_w,
                                      conv_b, x_proj_w, dt_proj_w, dt_proj_b,
                                      z_ws, xc_ws, dt_ws, Bm_ws, Cm_ws);
  mamba_scan<<<B_, 256, 0, stream>>>(xc_ws, dt_ws, z_ws, Bm_ws, Cm_ws,
                                     A_log, Dp, pooled);
  mamba_out<<<1, 128, 0, stream>>>(pooled, out_proj_w, (float*)d_out);
}
